// Model_36610301231456
// MI455X (gfx1250) — matrix operand order repaired and checked by reading, not compiled, not run on hardware
//
#include <hip/hip_runtime.h>
#include <hip/hip_bf16.h>
#include <stdint.h>

// ---- model dims ----
#define B_     32
#define SEQ_   512
#define NVAR_  512
#define NMARK_ 4
#define T_     516          // NVAR + NMARK
#define M_     (B_*T_)      // 16512 rows, = 516 * 32
#define DM_    512
#define DS_    32
#define DTR_   32
#define PL_    96
#define ZD_    64

typedef __bf16 bf16_t;
typedef __attribute__((ext_vector_type(16))) __bf16 bf16x16;
typedef __attribute__((ext_vector_type(8)))  float  f32x8;

// ---------------- per-(b,var) mean / std over SEQ ----------------
__global__ void k_stats(const float* __restrict__ x, float* mean, float* stdv) {
  int wave = (int)((blockIdx.x * blockDim.x + threadIdx.x) >> 5);
  int lane = threadIdx.x & 31;
  if (wave >= B_ * NVAR_) return;
  int b = wave / NVAR_, v = wave % NVAR_;
  const float* base = x + (size_t)b * SEQ_ * NVAR_ + v;
  float s = 0.f, sq = 0.f;
  for (int i = lane; i < SEQ_; i += 32) {
    float t = base[(size_t)i * NVAR_];
    s += t; sq += t * t;
  }
  for (int off = 16; off; off >>= 1) {
    s  += __shfl_xor(s,  off, 32);
    sq += __shfl_xor(sq, off, 32);
  }
  if (lane == 0) {
    float m   = s / (float)SEQ_;
    float var = sq / (float)SEQ_ - m * m;
    mean[wave] = m;
    stdv[wave] = sqrtf(var + 1e-5f);
  }
}

// ---------------- build token matrix (M x 512) in bf16 ----------------
__global__ void k_tok(const float* __restrict__ xe, const float* __restrict__ xm,
                      const float* __restrict__ mean, const float* __restrict__ stdv,
                      bf16_t* __restrict__ tok) {
  int idx = blockIdx.x * blockDim.x + threadIdx.x;
  if (idx >= M_ * SEQ_) return;
  int s   = idx & (SEQ_ - 1);
  int row = idx >> 9;
  int b = row / T_, tk = row % T_;
  float v;
  if (tk < NVAR_)
    v = (xe[((size_t)b * SEQ_ + s) * NVAR_ + tk] - mean[b * NVAR_ + tk]) / stdv[b * NVAR_ + tk];
  else
    v = xm[((size_t)b * SEQ_ + s) * NMARK_ + (tk - NVAR_)];
  tok[idx] = (bf16_t)v;
}

// ---------------- weight convert: W[K][N] f32 -> Wt[N][K] bf16 ----------------
__global__ void k_wT(const float* __restrict__ W, bf16_t* __restrict__ Wt, int K, int N) {
  int idx = blockIdx.x * blockDim.x + threadIdx.x;
  if (idx >= K * N) return;
  int k = idx / N, n = idx % N;
  Wt[(size_t)n * K + k] = (bf16_t)W[idx];
}

// ---------------- f32 -> bf16 row convert, optional time-reverse ----------------
__global__ void k_cvt(const float* __restrict__ src, int lds, int cols,
                      bf16_t* __restrict__ dst, int ldd, int rev) {
  int idx = blockIdx.x * blockDim.x + threadIdx.x;
  if (idx >= M_ * cols) return;
  int r = idx / cols, c = idx % cols;
  int sr = r;
  if (rev) { int b = r / T_, t = r % T_; sr = b * T_ + (T_ - 1 - t); }
  dst[(size_t)r * ldd + c] = (bf16_t)src[(size_t)sr * lds + c];
}

// ---------------- WMMA bf16 GEMM, 2x2 16x16 tiles per wave (no spills) ----------
// C(MxN,f32) = A(MxK bf16) * Wt(NxK bf16)^T + bias.  act: 0 none, 1 relu, 2 softplus.
__global__ void k_gemm(const bf16_t* __restrict__ A, int lda,
                       const bf16_t* __restrict__ Wt,
                       const float* __restrict__ bias,
                       float* __restrict__ C, int ldc,
                       bf16_t* __restrict__ Cb,
                       int N, int K, int act) {
  const int ntm = M_ / 32;            // 516 super-tiles of 32 rows
  int wave = (int)((blockIdx.x * blockDim.x + threadIdx.x) >> 5);
  int lane = threadIdx.x & 31;
  int ntn  = N >> 5;                  // N is a multiple of 32 for every GEMM here
  if (wave >= ntm * ntn) return;
  int tm = wave % ntm, tn = wave / ntm;
  int m0 = tm << 5, n0 = tn << 5;
  int h   = lane >> 4;                // half-wave selector
  int l15 = lane & 15;

  f32x8 acc00, acc01, acc10, acc11;
#pragma unroll
  for (int r = 0; r < 8; ++r) { acc00[r] = 0.f; acc01[r] = 0.f; acc10[r] = 0.f; acc11[r] = 0.f; }

  const bf16_t* arow0 = A  + (size_t)(m0 + l15)      * lda;
  const bf16_t* arow1 = A  + (size_t)(m0 + 16 + l15) * lda;
  const bf16_t* bcol0 = Wt + (size_t)(n0 + l15)      * K + h * 16;
  const bf16_t* bcol1 = Wt + (size_t)(n0 + 16 + l15) * K + h * 16;

  for (int k0 = 0; k0 < K; k0 += 32) {
    union { uint32_t u[8]; bf16x16 v; } ua0, ua1, ub0, ub1;
#pragma unroll
    for (int p = 0; p < 8; ++p) {
      // A 16x32: lanes 0-15 K=0..15, lanes 16-31 K=16..31 (contiguous, matches B)
      int kk = k0 + h * 16 + 2 * p;
      ua0.u[p] = *(const uint32_t*)(arow0 + kk);
      ua1.u[p] = *(const uint32_t*)(arow1 + kk);
    }
#pragma unroll
    for (int p = 0; p < 8; ++p) {
      // B 32x16: lane = column, lanes 0-15 K=0..15, lanes 16-31 K=16..31 (pairs contiguous)
      ub0.u[p] = *(const uint32_t*)(bcol0 + k0 + 2 * p);
      ub1.u[p] = *(const uint32_t*)(bcol1 + k0 + 2 * p);
    }
    __builtin_prefetch(arow0 + k0 + 32, 0, 1);   // global_prefetch_b8
    __builtin_prefetch(arow1 + k0 + 32, 0, 1);
    acc00 = __builtin_amdgcn_wmma_f32_16x16x32_bf16(false, ua0.v, false, ub0.v, (short)0, acc00, false, false);
    acc01 = __builtin_amdgcn_wmma_f32_16x16x32_bf16(false, ua0.v, false, ub1.v, (short)0, acc01, false, false);
    acc10 = __builtin_amdgcn_wmma_f32_16x16x32_bf16(false, ua1.v, false, ub0.v, (short)0, acc10, false, false);
    acc11 = __builtin_amdgcn_wmma_f32_16x16x32_bf16(false, ua1.v, false, ub1.v, (short)0, acc11, false, false);
  }

  int nn0 = n0 + l15, nn1 = n0 + 16 + l15;
  float bb0 = bias ? bias[nn0] : 0.f;
  float bb1 = bias ? bias[nn1] : 0.f;
#pragma unroll
  for (int r = 0; r < 8; ++r) {
    int mm0 = m0 + h * 8 + r;           // C/D layout: VGPR r -> row r (+8 upper half)
    int mm1 = mm0 + 16;
    float v00 = acc00[r] + bb0, v01 = acc01[r] + bb1;
    float v10 = acc10[r] + bb0, v11 = acc11[r] + bb1;
    if (act == 1) {
      v00 = v00 > 0.f ? v00 : 0.f; v01 = v01 > 0.f ? v01 : 0.f;
      v10 = v10 > 0.f ? v10 : 0.f; v11 = v11 > 0.f ? v11 : 0.f;
    } else if (act == 2) {
      v00 = (v00 > 20.f) ? v00 : logf(1.f + __expf(v00));
      v01 = (v01 > 20.f) ? v01 : logf(1.f + __expf(v01));
      v10 = (v10 > 20.f) ? v10 : logf(1.f + __expf(v10));
      v11 = (v11 > 20.f) ? v11 : logf(1.f + __expf(v11));
    }
    C[(size_t)mm0 * ldc + nn0] = v00;  C[(size_t)mm0 * ldc + nn1] = v01;
    C[(size_t)mm1 * ldc + nn0] = v10;  C[(size_t)mm1 * ldc + nn1] = v11;
    if (Cb) {
      Cb[(size_t)mm0 * ldc + nn0] = (bf16_t)v00;  Cb[(size_t)mm0 * ldc + nn1] = (bf16_t)v01;
      Cb[(size_t)mm1 * ldc + nn0] = (bf16_t)v10;  Cb[(size_t)mm1 * ldc + nn1] = (bf16_t)v11;
    }
  }
}

// ---------------- depthwise conv(DCONV=2) + silu; xi = xz[:, :512] ----------------
__global__ void k_conv(const float* __restrict__ xz, const float* __restrict__ cw,
                       const float* __restrict__ cb, float* __restrict__ xc,
                       bf16_t* __restrict__ xcb) {
  int idx = blockIdx.x * blockDim.x + threadIdx.x;
  if (idx >= M_ * DM_) return;
  int r = idx >> 9, d = idx & 511;
  int t = r % T_;
  float cur  = xz[(size_t)r * 1024 + d];
  float prev = (t > 0) ? xz[(size_t)(r - 1) * 1024 + d] : 0.f;
  float a = prev * cw[d * 2 + 0] + cur * cw[d * 2 + 1] + cb[d];
  float s = a / (1.f + __expf(-a));
  xc[idx]  = s;
  xcb[idx] = (bf16_t)s;
}

__global__ void k_aneg(const float* __restrict__ alog, float* __restrict__ an) {
  int idx = blockIdx.x * blockDim.x + threadIdx.x;
  if (idx >= DM_ * DS_) return;
  an[idx] = -__expf(alog[idx]);
}

// ---------------- selective scan: one wave per (b, channel), lane = state ----------------
__global__ void k_scan(const float* __restrict__ dt, const float* __restrict__ xdbl,
                       const float* __restrict__ xc, const float* __restrict__ an,
                       const float* __restrict__ Dv, const float* __restrict__ xz,
                       float* __restrict__ y) {
  int wave = (int)((blockIdx.x * blockDim.x + threadIdx.x) >> 5);
  int lane = threadIdx.x & 31;
  if (wave >= B_ * DM_) return;
  int b = wave / DM_, d = wave % DM_;
  float As = an[d * DS_ + lane];
  float Dd = Dv[d];
  float h = 0.f;
  size_t row = (size_t)b * T_;
  for (int t = 0; t < T_; ++t, ++row) {
    float dt_t = dt[row * DM_ + d];
    float x_t  = xc[row * DM_ + d];
    float bs   = xdbl[row * 96 + 32 + lane];
    float cs   = xdbl[row * 96 + 64 + lane];
    h = h * __expf(dt_t * As) + dt_t * x_t * bs;
    float ys = h * cs;
#pragma unroll
    for (int off = 16; off; off >>= 1) ys += __shfl_xor(ys, off, 32);
    if (lane == 0) {
      float zv = xz[row * 1024 + 512 + d];
      y[row * DM_ + d] = (ys + x_t * Dd) * (zv / (1.f + __expf(-zv)));
    }
  }
}

// ---------------- residual adds ----------------
__global__ void k_add3(float* __restrict__ x, const float* __restrict__ a,
                       const float* __restrict__ brev) {
  int idx = blockIdx.x * blockDim.x + threadIdx.x;
  if (idx >= M_ * DM_) return;
  int r = idx >> 9, c = idx & 511;
  int b = r / T_, t = r % T_;
  size_t rr = ((size_t)b * T_ + (T_ - 1 - t)) * DM_ + c;
  x[idx] += a[idx] + brev[rr];
}

__global__ void k_add2(float* __restrict__ x, const float* __restrict__ a) {
  int idx = blockIdx.x * blockDim.x + threadIdx.x;
  if (idx >= M_ * DM_) return;
  x[idx] += a[idx];
}

// ---------------- layernorm over 512 cols; writes f32 + bf16 ----------------
__global__ void k_ln(const float* __restrict__ src, const float* __restrict__ g,
                     const float* __restrict__ be, float* __restrict__ out,
                     bf16_t* __restrict__ outb) {
  int r = blockIdx.x, tid = threadIdx.x;     // 256 threads, 2 cols each
  float v0 = src[(size_t)r * 512 + tid];
  float v1 = src[(size_t)r * 512 + tid + 256];
  float s = v0 + v1, sq = v0 * v0 + v1 * v1;
  for (int off = 16; off; off >>= 1) {
    s  += __shfl_xor(s,  off, 32);
    sq += __shfl_xor(sq, off, 32);
  }
  __shared__ float ss[8], qq[8];
  int w = tid >> 5;
  if ((tid & 31) == 0) { ss[w] = s; qq[w] = sq; }
  __syncthreads();
  if (tid == 0) {
    float a = 0.f, b2 = 0.f;
    for (int i = 0; i < 8; ++i) { a += ss[i]; b2 += qq[i]; }
    ss[0] = a; qq[0] = b2;
  }
  __syncthreads();
  float mean = ss[0] / 512.f;
  float var  = qq[0] / 512.f - mean * mean;
  float rstd = rsqrtf(var + 1e-5f);
  float o0 = (v0 - mean) * rstd * g[tid]       + be[tid];
  float o1 = (v1 - mean) * rstd * g[tid + 256] + be[tid + 256];
  out[(size_t)r * 512 + tid]        = o0;
  out[(size_t)r * 512 + tid + 256]  = o1;
  outb[(size_t)r * 512 + tid]       = (bf16_t)o0;
  outb[(size_t)r * 512 + tid + 256] = (bf16_t)o1;
}

// ---------------- reparameterization -> bf16 ----------------
__global__ void k_z(const float* __restrict__ mu, const float* __restrict__ lv,
                    const float* __restrict__ eps, bf16_t* __restrict__ zb) {
  int idx = blockIdx.x * blockDim.x + threadIdx.x;
  if (idx >= M_ * ZD_) return;
  zb[idx] = (bf16_t)(mu[idx] + __expf(0.5f * lv[idx]) * eps[idx]);
}

// ---------------- de-normalize + transpose to (B, PL, NVAR) ----------------
__global__ void k_out(const float* __restrict__ decf, const float* __restrict__ mean,
                      const float* __restrict__ stdv, float* __restrict__ out) {
  int idx = blockIdx.x * blockDim.x + threadIdx.x;
  if (idx >= B_ * PL_ * NVAR_) return;
  int v = idx % NVAR_;
  int p = (idx / NVAR_) % PL_;
  int b = idx / (NVAR_ * PL_);
  size_t row = (size_t)b * T_ + v;
  out[idx] = decf[row * PL_ + p] * stdv[b * NVAR_ + v] + mean[b * NVAR_ + v];
}

// =====================================================================
extern "C" void kernel_launch(void* const* d_in, const int* in_sizes, int n_in,
                              void* d_out, int out_size, void* d_ws, size_t ws_size,
                              hipStream_t stream) {
  (void)in_sizes; (void)n_in; (void)out_size; (void)ws_size;

  const float* x_enc  = (const float*)d_in[0];
  const float* x_mark = (const float*)d_in[1];
  const float* eps    = (const float*)d_in[4];
  auto P = [&](int i) { return (const float*)d_in[i]; };
  // flattened param order: layers[l] at 5+26l:
  //   m: +0 in_proj,+1 conv_w,+2 conv_b,+3 x_proj,+4 dt_proj,+5 dt_bias,+6 A_log,+7 D,+8 out_proj
  //   mr: +9..+17, then +18 w1,+19 b1,+20 w2,+21 b2,+22 ln1_g,+23 ln1_b,+24 ln2_g,+25 ln2_b
  // 57 emb_w, 58 emb_b, 59 norm_g, 60 norm_b, 61 mu_w, 62 mu_b, 63 lv_w, 64 lv_b, 65 p2_w, 66 p2_b

  // ---- workspace carve ----
  char* ws = (char*)d_ws;
  size_t off = 0;
  auto carve = [&](size_t bytes) -> void* {
    void* p = ws + off;
    off += (bytes + 255) & ~(size_t)255;
    return p;
  };
  float*  MEAN = (float*)carve((size_t)B_ * NVAR_ * 4);
  float*  STDV = (float*)carve((size_t)B_ * NVAR_ * 4);
  float*  X    = (float*)carve((size_t)M_ * DM_ * 4);
  bf16_t* XB   = (bf16_t*)carve((size_t)M_ * DM_ * 2);
  bf16_t* XB2  = (bf16_t*)carve((size_t)M_ * DM_ * 2);
  float*  S0   = (float*)carve((size_t)M_ * 1024 * 4);   // xz
  float*  S1   = (float*)carve((size_t)M_ * DM_ * 4);    // xc
  float*  S2   = (float*)carve((size_t)M_ * DM_ * 4);    // dt / ffn2
  float*  S4   = (float*)carve((size_t)M_ * DM_ * 4);    // fwd mamba out
  float*  S5   = (float*)carve((size_t)M_ * DM_ * 4);    // rev mamba out
  float*  XDBL = (float*)carve((size_t)M_ * 96 * 4);
  bf16_t* DTB  = (bf16_t*)carve((size_t)M_ * DTR_ * 2);
  float*  MU   = (float*)carve((size_t)M_ * ZD_ * 4);
  float*  LV   = (float*)carve((size_t)M_ * ZD_ * 4);
  bf16_t* ZB   = (bf16_t*)carve((size_t)M_ * ZD_ * 2);
  float*  DECF = (float*)carve((size_t)M_ * PL_ * 4);
  float*  ANEG = (float*)carve((size_t)DM_ * DS_ * 4);
  bf16_t* WT   = (bf16_t*)carve((size_t)1024 * 512 * 2); // shared weight staging

  auto blks = [](long n) { return (unsigned)((n + 255) / 256); };

  auto gemm = [&](const bf16_t* A, int lda, const float* W, int K, int N,
                  const float* bias, float* C, int ldc, bf16_t* Cb, int act) {
    k_wT<<<blks((long)K * N), 256, 0, stream>>>(W, WT, K, N);
    int nwaves = (M_ / 32) * (N / 32);
    k_gemm<<<(unsigned)((nwaves + 7) / 8), 256, 0, stream>>>(
        A, lda, WT, bias, C, ldc, Cb, N, K, act);
  };

  // ---- stats + tokens + embedding ----
  k_stats<<<(B_ * NVAR_) / 8, 256, 0, stream>>>(x_enc, MEAN, STDV);
  k_tok<<<blks((long)M_ * SEQ_), 256, 0, stream>>>(x_enc, x_mark, MEAN, STDV, XB);
  gemm(XB, SEQ_, P(57), SEQ_, DM_, P(58), X, DM_, nullptr, 0);

  auto mamba = [&](int pb, int rev, float* outbuf) {
    k_cvt<<<blks((long)M_ * DM_), 256, 0, stream>>>(X, DM_, DM_, XB, DM_, rev);
    gemm(XB, DM_, P(pb + 0), DM_, 1024, nullptr, S0, 1024, nullptr, 0);           // in_proj
    k_conv<<<blks((long)M_ * DM_), 256, 0, stream>>>(S0, P(pb + 1), P(pb + 2), S1, XB);
    gemm(XB, DM_, P(pb + 3), DM_, 96, nullptr, XDBL, 96, nullptr, 0);             // x_proj
    k_cvt<<<blks((long)M_ * DTR_), 256, 0, stream>>>(XDBL, 96, DTR_, DTB, DTR_, 0);
    gemm(DTB, DTR_, P(pb + 4), DTR_, DM_, P(pb + 5), S2, DM_, nullptr, 2);        // dt_proj + softplus
    k_aneg<<<blks(DM_ * DS_), 256, 0, stream>>>(P(pb + 6), ANEG);
    k_scan<<<(B_ * DM_) / 8, 256, 0, stream>>>(S2, XDBL, S1, ANEG, P(pb + 7), S0, outbuf);
    k_cvt<<<blks((long)M_ * DM_), 256, 0, stream>>>(outbuf, DM_, DM_, XB, DM_, 0);
    gemm(XB, DM_, P(pb + 8), DM_, DM_, nullptr, outbuf, DM_, nullptr, 0);         // out_proj
  };

  for (int l = 0; l < 2; ++l) {
    int lb = 5 + l * 26;
    mamba(lb + 0, 0, S4);
    mamba(lb + 9, 1, S5);
    k_add3<<<blks((long)M_ * DM_), 256, 0, stream>>>(X, S4, S5);
    k_ln<<<M_, 256, 0, stream>>>(X, P(lb + 22), P(lb + 23), X, XB);
    gemm(XB, DM_, P(lb + 18), DM_, DM_, P(lb + 19), S1, DM_, XB2, 1);             // w1 + relu
    gemm(XB2, DM_, P(lb + 20), DM_, DM_, P(lb + 21), S2, DM_, nullptr, 0);        // w2
    k_add2<<<blks((long)M_ * DM_), 256, 0, stream>>>(X, S2);
    k_ln<<<M_, 256, 0, stream>>>(X, P(lb + 24), P(lb + 25), X, XB);
  }

  // ---- head ----
  k_ln<<<M_, 256, 0, stream>>>(X, P(59), P(60), S1, XB);
  gemm(XB, DM_, P(61), DM_, ZD_, P(62), MU, ZD_, nullptr, 0);
  gemm(XB, DM_, P(63), DM_, ZD_, P(64), LV, ZD_, nullptr, 0);
  k_z<<<blks((long)M_ * ZD_), 256, 0, stream>>>(MU, LV, eps, ZB);
  gemm(ZB, ZD_, P(65), ZD_, PL_, P(66), DECF, PL_, nullptr, 0);
  k_out<<<blks((long)B_ * PL_ * NVAR_), 256, 0, stream>>>(DECF, MEAN, STDV, (float*)d_out);
}
